// HVFeatureNetFinal_41326175322617
// MI455X (gfx1250) — matrix operand order repaired and checked by reading, not compiled, not run on hardware
//
#include <hip/hip_runtime.h>

typedef __attribute__((ext_vector_type(16))) __bf16    v16bf;
typedef __attribute__((ext_vector_type(8)))  float     v8f;
typedef __attribute__((ext_vector_type(4)))  unsigned  uv4;
typedef __attribute__((ext_vector_type(4)))  int       iv4;
typedef __attribute__((ext_vector_type(8)))  int       iv8;

#define NPTS   200000
#define NTILE  12500      // NPTS / 16

// ---- scale constants ----
// V:       40000, 20000, 10000, 20000, 10000
// sum cum: 0, 40000, 60000, 70000, 90000  (x7 floats)
// vf012 (u32, 64ch): 0, 2560000, 3840000  (total 4480000)
// vf34  (u32, 128ch): scale3 @0 (2560000), scale4 @2560000 (1280000)
#define OUT3_ELEMS 9048064ull   // 2*128*188*188
#define OUT_TOTAL  11310080ull  // + 2*128*94*94

// ---- workspace layout (bytes) ----
constexpr size_t OFF_SUMS  = 0;                                      // 700000 f32
constexpr size_t OFF_VF012 = (700000ull * 4 + 255) & ~255ull;        // 4480000 u32
constexpr size_t OFF_VF34  = OFF_VF012 + 4480000ull * 4;             // 3840000 u32
constexpr size_t OFF_FEAT  = OFF_VF34  + 3840000ull * 4;             // 76800000 bf16
constexpr size_t OFF_WPO   = OFF_FEAT  + 76800000ull * 2;            // 49152 bf16
constexpr size_t OFF_WAO   = OFF_WPO   + 49152ull * 2;               // 4096 bf16 (contiguous after WPO)
constexpr size_t OFF_WPA   = OFF_WAO   + 4096ull * 2;                // 2048 bf16
constexpr size_t OFF_WAA   = OFF_WPA   + 2048ull * 2;                // 2048 bf16 (contiguous after WPA)

#if defined(__has_builtin)
#if __has_builtin(__builtin_amdgcn_tensor_load_to_lds) && __has_builtin(__builtin_amdgcn_s_wait_tensorcnt)
#define HAVE_TDM 1
#endif
#endif

__device__ __forceinline__ unsigned short f2bf(float f) {
  unsigned u = __float_as_uint(f);
  unsigned r = u + 0x7FFFu + ((u >> 16) & 1u);   // round-to-nearest-even
  return (unsigned short)(r >> 16);
}
// order-preserving float <-> uint map so scatter-max uses native atomic_max_u32
__device__ __forceinline__ unsigned encf(float f) {
  unsigned u = __float_as_uint(f);
  return (u & 0x80000000u) ? ~u : (u | 0x80000000u);
}
__device__ __forceinline__ float decf(unsigned k) {
  unsigned u = (k & 0x80000000u) ? (k ^ 0x80000000u) : ~k;
  return __uint_as_float(u);
}
#define ENC_FMIN 0x00800000u   // encf(-3.4028235e38f)

union FragBF {
  v16bf v;
  unsigned short h[16];
  uv4 u[2];
};

struct PtFeats { float pif[9]; float att[18]; };

__device__ __forceinline__ PtFeats point_feats(const float* pts, const float* vlc,
                                               const int* pv, const float* sums,
                                               int scum, int p) {
  PtFeats r;
  float4 pt = ((const float4*)pts)[p];
  float vl0 = vlc[p * 3 + 0], vl1 = vlc[p * 3 + 1];
  int v = pv[p];
  const float* sp = sums + (size_t)(scum + v) * 7;
  float inv = 1.0f / fmaxf(sp[6], 1.0f);
  float pm0 = sp[0] * inv, pm1 = sp[1] * inv, pm2 = sp[2] * inv, pm3 = sp[3] * inv;
  float mv0 = sp[4] * inv, mv1 = sp[5] * inv;
  float l0 = pt.x - pm0, l1 = pt.y - pm1, l2 = pt.z - pm2;
  r.pif[0]=pt.x; r.pif[1]=pt.y; r.pif[2]=pt.z; r.pif[3]=pt.w;
  r.pif[4]=l0;   r.pif[5]=l1;   r.pif[6]=l2;   r.pif[7]=vl0; r.pif[8]=vl1;
  r.att[0]=l0;  r.att[1]=l1;  r.att[2]=l2;  r.att[3]=pt.w;
  r.att[4]=l0;  r.att[5]=l1;  r.att[6]=l2;  r.att[7]=vl0; r.att[8]=vl1;
  r.att[9]=pm0; r.att[10]=pm1; r.att[11]=pm2; r.att[12]=pm3;
  r.att[13]=0.f; r.att[14]=0.f; r.att[15]=0.f; r.att[16]=mv0; r.att[17]=mv1;
  return r;
}

// A-fragment (16x32 bf16): lane l -> row l&15; K(j) = j + (j<8?0:8) + (l<16?0:8)
__device__ __forceinline__ FragBF frag_from9(const float x[9], int lane) {
  FragBF f;
  if (lane < 16) {
#pragma unroll
    for (int j = 0; j < 8; ++j) f.h[j] = f2bf(x[j]);
#pragma unroll
    for (int j = 8; j < 16; ++j) f.h[j] = 0;
  } else {
    f.h[0] = f2bf(x[8]);
#pragma unroll
    for (int j = 1; j < 16; ++j) f.h[j] = 0;
  }
  return f;
}
__device__ __forceinline__ FragBF frag_from18(const float x[18], int lane) {
  FragBF f;
  if (lane < 16) {
#pragma unroll
    for (int j = 0; j < 8; ++j) f.h[j] = f2bf(x[j]);     // K=0..7
    f.h[8] = f2bf(x[16]); f.h[9] = f2bf(x[17]);          // K=16,17
#pragma unroll
    for (int j = 10; j < 16; ++j) f.h[j] = 0;
  } else {
#pragma unroll
    for (int j = 0; j < 8; ++j) f.h[j] = f2bf(x[8 + j]); // K=8..15
#pragma unroll
    for (int j = 8; j < 16; ++j) f.h[j] = 0;             // K=24..31
  }
  return f;
}

#ifdef HAVE_TDM
// 1-D contiguous TDM copy, data_size = 8B, nelem8 elements (< 65536).
__device__ __forceinline__ void tdm_load_1d(const void* gsrc, unsigned lds_byte_addr,
                                            unsigned nelem8) {
  unsigned long long ga = (unsigned long long)gsrc;
  uv4 g0 = { 1u,                                   // count=1, user mode, no gather
             lds_byte_addr,                        // D#[63:32]  lds_addr
             (unsigned)(ga & 0xFFFFFFFFu),         // D#[95:64]  global_addr lo
             (unsigned)((ga >> 32) & 0x01FFFFFFu) | (2u << 30) };  // ga hi | type=2
  iv8 g1 = { (int)(3u << 16),                      // data_size=8B, wg_mask=0
             (int)((nelem8 & 0xFFFFu) << 16),      // tensor_dim0[15:0]
             (int)(((nelem8 >> 16) & 0xFFFFu) | (1u << 16)),  // td0 hi | tensor_dim1=1
             (int)((nelem8 & 0xFFFFu) << 16),      // td1 hi(0) | tile_dim0
             0,                                    // tile_dim1=0, tile_dim2=0
             (int)nelem8,                          // tensor_dim0_stride lo
             0, 0 };
  iv4 z4 = {0, 0, 0, 0};
#if __clang_major__ >= 23
  iv8 z8 = {0, 0, 0, 0, 0, 0, 0, 0};
  __builtin_amdgcn_tensor_load_to_lds(g0, g1, z4, z4, z8, 0);
#else
  __builtin_amdgcn_tensor_load_to_lds(g0, g1, z4, z4, 0);
#endif
}
#endif

// ---------------- kernel 0: init ----------------
__global__ void k_init(float* out, float* sums, unsigned* vf012, unsigned* vf34) {
  size_t i = (size_t)blockIdx.x * blockDim.x + threadIdx.x;
  size_t stride = (size_t)gridDim.x * blockDim.x;
  const size_t T0 = OUT_TOTAL, T1 = T0 + 700000ull, T2 = T1 + 4480000ull, T3 = T2 + 3840000ull;
  for (size_t x = i; x < T3; x += stride) {
    if (x < T0)      out[x] = 0.f;
    else if (x < T1) sums[x - T0] = 0.f;
    else if (x < T2) vf012[x - T1] = ENC_FMIN;
    else             vf34[x - T2] = ENC_FMIN;
  }
}

// ---------------- kernel 1: per-voxel sums ----------------
__global__ void k_accum(const float* __restrict__ pts,
                        const float* vlc0, const float* vlc1, const float* vlc2,
                        const float* vlc3, const float* vlc4,
                        const int* pv0, const int* pv1, const int* pv2,
                        const int* pv3, const int* pv4, float* sums) {
  int n = blockIdx.x * blockDim.x + threadIdx.x;
  if (n >= NPTS) return;
  float4 p = ((const float4*)pts)[n];
  const float* vlcs[5] = {vlc0, vlc1, vlc2, vlc3, vlc4};
  const int*   pvs[5]  = {pv0, pv1, pv2, pv3, pv4};
  const int    cum[5]  = {0, 40000, 60000, 70000, 90000};
#pragma unroll
  for (int s = 0; s < 5; ++s) {
    int v = pvs[s][n];
    float a = vlcs[s][n * 3 + 0], b = vlcs[s][n * 3 + 1];
    float* base = sums + (size_t)(cum[s] + v) * 7;
    atomicAdd(base + 0, p.x); atomicAdd(base + 1, p.y);
    atomicAdd(base + 2, p.z); atomicAdd(base + 3, p.w);
    atomicAdd(base + 4, a);   atomicAdd(base + 5, b);
    atomicAdd(base + 6, 1.0f);
  }
}

// ---------------- kernel 2: swizzle weights to B-fragment layout ----------------
// B (32x16 bf16): lane l -> col l&15; K(j) = j + (j<8?0:8) + (l<16?0:8); chunk adds 32*c
__global__ void k_prepw(const float* Wpa, const float* Waa, const float* Wpo, const float* Wao,
                        unsigned short* sWpA, unsigned short* sWaA,
                        unsigned short* sWpO, unsigned short* sWaO) {
  int id = blockIdx.x * blockDim.x + threadIdx.x;
  if (id < 49152) {                       // WpO: K=384(12 chunks), N=128(8 tiles)
    int j = id & 15, l = (id >> 4) & 31, t = (id >> 9) & 7, c = id >> 12;
    int K = 32 * c + j + ((j < 8) ? 0 : 8) + ((l < 16) ? 0 : 8);
    int col = 16 * t + (l & 15);
    sWpO[id] = f2bf(Wpo[K * 128 + col]);
  } else if (id < 53248) {                // WaO: K=18 (pad 32), N=128
    int i2 = id - 49152;
    int j = i2 & 15, l = (i2 >> 4) & 31, t = (i2 >> 9) & 7;
    int K = j + ((j < 8) ? 0 : 8) + ((l < 16) ? 0 : 8), col = 16 * t + (l & 15);
    sWaO[i2] = (K < 18) ? f2bf(Wao[K * 128 + col]) : (unsigned short)0;
  } else if (id < 55296) {                // WpA: K=9 (pad 32), N=64
    int i2 = id - 53248;
    int j = i2 & 15, l = (i2 >> 4) & 31, t = (i2 >> 9) & 3;
    int K = j + ((j < 8) ? 0 : 8) + ((l < 16) ? 0 : 8), col = 16 * t + (l & 15);
    sWpA[i2] = (K < 9) ? f2bf(Wpa[K * 64 + col]) : (unsigned short)0;
  } else if (id < 57344) {                // WaA: K=18 (pad 32), N=64
    int i2 = id - 55296;
    int j = i2 & 15, l = (i2 >> 4) & 31, t = (i2 >> 9) & 3;
    int K = j + ((j < 8) ? 0 : 8) + ((l < 16) ? 0 : 8), col = 16 * t + (l & 15);
    sWaA[i2] = (K < 18) ? f2bf(Waa[K * 64 + col]) : (unsigned short)0;
  }
}

// ---------------- kernel 3: AVFE scales 0..2 (WMMA, LDS-staged weights) ----------------
// 8 waves/block, 4 (s,tile) jobs per wave.
__global__ void __launch_bounds__(256) k_avfe(const float* pts,
    const float* vlc0, const float* vlc1, const float* vlc2,
    const int* pv0, const int* pv1, const int* pv2,
    const float* sums, unsigned* vf012, unsigned short* feat,
    const unsigned short* sWpA /* WpA(2048) + WaA(2048) contiguous */) {
  __shared__ unsigned short ldsA[4096];   // 8 KB
  for (int k = threadIdx.x; k < 512; k += 256)
    ((uv4*)ldsA)[k] = ((const uv4*)sWpA)[k];
  __syncthreads();

  int w = threadIdx.x >> 5, lane = threadIdx.x & 31;
  for (int it = 0; it < 4; ++it) {
    int job = blockIdx.x * 32 + it * 8 + w;       // wave-uniform
    if (job >= 3 * NTILE) continue;
    int s = job % 3, tile = job / 3;
    const float* vlc = (s == 0) ? vlc0 : (s == 1) ? vlc1 : vlc2;
    const int*   pv  = (s == 0) ? pv0  : (s == 1) ? pv1  : pv2;
    const int scum   = (s == 0) ? 0 : (s == 1) ? 40000 : 60000;
    unsigned* vfb = vf012 + ((s == 0) ? 0 : (s == 1) ? 2560000 : 3840000);

    int row = lane & 15, p = tile * 16 + row;
    PtFeats F = point_feats(pts, vlc, pv, sums, scum, p);
    FragBF fp = frag_from9(F.pif, lane);
    FragBF fa = frag_from18(F.att, lane);
    int off = (lane < 16) ? 0 : 8;
    int pvr[8];
#pragma unroll
    for (int r = 0; r < 8; ++r) pvr[r] = pv[tile * 16 + r + off];

#pragma unroll
    for (int t = 0; t < 4; ++t) {
      FragBF bp, ba;
      const uv4* q = (const uv4*)(ldsA + (size_t)(t * 32 + lane) * 16);
      bp.u[0] = q[0]; bp.u[1] = q[1];
      q = (const uv4*)(ldsA + 2048 + (size_t)(t * 32 + lane) * 16);
      ba.u[0] = q[0]; ba.u[1] = q[1];
      v8f z = {0.f, 0.f, 0.f, 0.f, 0.f, 0.f, 0.f, 0.f};
      v8f pfc = __builtin_amdgcn_wmma_f32_16x16x32_bf16(false, fp.v, false, bp.v, (short)0, z, false, false);
      v8f afc = __builtin_amdgcn_wmma_f32_16x16x32_bf16(false, fa.v, false, ba.v, (short)0, z, false, false);
      v8f fin = pfc * afc;
#pragma unroll
      for (int r = 0; r < 8; ++r) {
        int M = r + off;
        float fv = fin[r];
        feat[(size_t)(tile * 16 + M) * 384 + s * 128 + t * 16 + (lane & 15)] = f2bf(fv);
        atomicMax(vfb + (size_t)pvr[r] * 64 + t * 16 + (lane & 15), encf(fv));
      }
    }
  }
}

// ---------------- kernel 4: gather voxel-max back to feat ----------------
__global__ void k_vfgather(const int* pv0, const int* pv1, const int* pv2,
                           const unsigned* vf012, unsigned short* feat) {
  int n = blockIdx.x * blockDim.x + threadIdx.x;
  if (n >= NPTS) return;
  const int* pvs[3] = {pv0, pv1, pv2};
  const int vc[3] = {0, 2560000, 3840000};
#pragma unroll
  for (int s = 0; s < 3; ++s) {
    const unsigned* vb = vf012 + vc[s] + (size_t)pvs[s][n] * 64;
    unsigned short* fb = feat + (size_t)n * 384 + s * 128 + 64;
    for (int c = 0; c < 64; ++c) fb[c] = f2bf(decf(vb[c]));
  }
}

// ---------------- kernel 5: fused 384x128 GEMM + scales 3,4 ----------------
// Weights (96KB + 8KB) staged to LDS once per workgroup via Tensor Data Mover;
// 8 waves/block, 4 tiles per wave.
__global__ void __launch_bounds__(256) k_main(const float* pts,
    const float* vlc3, const float* vlc4, const int* pv3, const int* pv4,
    const float* sums, const unsigned short* feat,
    const unsigned short* sWpO /* WpO(49152) + WaO(4096) contiguous */,
    unsigned* vf34) {
  __shared__ unsigned short ldsW[53248];  // 104 KB: WpO frags + WaO frags
#ifdef HAVE_TDM
  if (threadIdx.x < 32) {                 // wave 0 drives the TDM
    tdm_load_1d(sWpO, (unsigned)(size_t)(void*)ldsW, 13312u /* 106496 B / 8 */);
    __builtin_amdgcn_s_wait_tensorcnt(0);
  }
#else
  for (int k = threadIdx.x; k < 6656; k += 256)
    ((uv4*)ldsW)[k] = ((const uv4*)sWpO)[k];
#endif
  __syncthreads();

  int w = threadIdx.x >> 5, lane = threadIdx.x & 31;
  int row = lane & 15;
  int koff = (lane < 16) ? 0 : 8;
  int off = (lane < 16) ? 0 : 8;

  for (int it = 0; it < 4; ++it) {
    int tile = blockIdx.x * 32 + it * 8 + w;      // wave-uniform
    if (tile >= NTILE) continue;
    size_t rb = (size_t)(tile * 16 + row) * 384;

    FragBF A[12];
#pragma unroll
    for (int c = 0; c < 12; ++c) {
      A[c].u[0] = *(const uv4*)(feat + rb + 32 * c + koff);
      A[c].u[1] = *(const uv4*)(feat + rb + 32 * c + 16 + koff);
    }

    v8f acc[8];
#pragma unroll
    for (int t = 0; t < 8; ++t) {
      v8f a = {0.f, 0.f, 0.f, 0.f, 0.f, 0.f, 0.f, 0.f};
#pragma unroll
      for (int c = 0; c < 12; ++c) {
        FragBF b;
        const uv4* q = (const uv4*)(ldsW + (size_t)((c * 8 + t) * 32 + lane) * 16);
        b.u[0] = q[0]; b.u[1] = q[1];
        a = __builtin_amdgcn_wmma_f32_16x16x32_bf16(false, A[c].v, false, b.v, (short)0, a, false, false);
      }
      acc[t] = a;
    }

#pragma unroll
    for (int s2 = 0; s2 < 2; ++s2) {
      const float* vlc = s2 ? vlc4 : vlc3;
      const int*   pv  = s2 ? pv4  : pv3;
      int scum = s2 ? 90000 : 70000;
      unsigned* vfb = vf34 + (s2 ? 2560000 : 0);
      PtFeats F = point_feats(pts, vlc, pv, sums, scum, tile * 16 + row);
      FragBF fa = frag_from18(F.att, lane);
      int pvr[8];
#pragma unroll
      for (int r = 0; r < 8; ++r) pvr[r] = pv[tile * 16 + r + off];
#pragma unroll
      for (int t = 0; t < 8; ++t) {
        FragBF ba;
        const uv4* q = (const uv4*)(ldsW + 49152 + (size_t)(t * 32 + lane) * 16);
        ba.u[0] = q[0]; ba.u[1] = q[1];
        v8f z = {0.f, 0.f, 0.f, 0.f, 0.f, 0.f, 0.f, 0.f};
        v8f afc = __builtin_amdgcn_wmma_f32_16x16x32_bf16(false, fa.v, false, ba.v, (short)0, z, false, false);
        v8f fin = afc * acc[t];
#pragma unroll
        for (int r = 0; r < 8; ++r)
          atomicMax(vfb + (size_t)pvr[r] * 128 + t * 16 + (lane & 15), encf(fin[r]));
      }
    }
  }
}

// ---------------- kernel 6: dense BEV scatter ----------------
__global__ void k_scatter(const unsigned* vf34, const int* vfs3, const int* vfs4, float* out) {
  int i = blockIdx.x * blockDim.x + threadIdx.x;
  if (i >= 3840000) return;
  if (i < 2560000) {
    int v = i >> 7, c = i & 127;
    int b = vfs3[v * 3], y = vfs3[v * 3 + 1], x = vfs3[v * 3 + 2];
    out[(((size_t)b * 128 + c) * 188 + y) * 188 + x] = decf(vf34[i]);
  } else {
    int j = i - 2560000;
    int v = j >> 7, c = j & 127;
    int b = vfs4[v * 3], y = vfs4[v * 3 + 1], x = vfs4[v * 3 + 2];
    out[OUT3_ELEMS + (((size_t)b * 128 + c) * 94 + y) * 94 + x] = decf(vf34[i]);
  }
}

extern "C" void kernel_launch(void* const* d_in, const int* in_sizes, int n_in,
                              void* d_out, int out_size, void* d_ws, size_t ws_size,
                              hipStream_t stream) {
  (void)in_sizes; (void)n_in; (void)out_size; (void)ws_size;
  const float* pts  = (const float*)d_in[0];
  const float* vlc0 = (const float*)d_in[1];
  const int*   pv0  = (const int*)d_in[2];
  const float* vlc1 = (const float*)d_in[4];
  const int*   pv1  = (const int*)d_in[5];
  const float* vlc2 = (const float*)d_in[7];
  const int*   pv2  = (const int*)d_in[8];
  const float* vlc3 = (const float*)d_in[10];
  const int*   pv3  = (const int*)d_in[11];
  const int*   vfs3 = (const int*)d_in[12];
  const float* vlc4 = (const float*)d_in[13];
  const int*   pv4  = (const int*)d_in[14];
  const int*   vfs4 = (const int*)d_in[15];
  const float* Wpa  = (const float*)d_in[16];
  const float* Waa  = (const float*)d_in[17];
  const float* Wpo  = (const float*)d_in[18];
  const float* Wao  = (const float*)d_in[19];

  char* ws = (char*)d_ws;
  float*          sums  = (float*)(ws + OFF_SUMS);
  unsigned*       vf012 = (unsigned*)(ws + OFF_VF012);
  unsigned*       vf34  = (unsigned*)(ws + OFF_VF34);
  unsigned short* feat  = (unsigned short*)(ws + OFF_FEAT);
  unsigned short* sWpO  = (unsigned short*)(ws + OFF_WPO);
  unsigned short* sWaO  = (unsigned short*)(ws + OFF_WAO);
  unsigned short* sWpA  = (unsigned short*)(ws + OFF_WPA);
  unsigned short* sWaA  = (unsigned short*)(ws + OFF_WAA);
  float* out = (float*)d_out;

  k_init<<<4096, 256, 0, stream>>>(out, sums, vf012, vf34);
  k_accum<<<(NPTS + 255) / 256, 256, 0, stream>>>(pts, vlc0, vlc1, vlc2, vlc3, vlc4,
                                                  pv0, pv1, pv2, pv3, pv4, sums);
  k_prepw<<<224, 256, 0, stream>>>(Wpa, Waa, Wpo, Wao, sWpA, sWaA, sWpO, sWaO);
  k_avfe<<<(3 * NTILE + 31) / 32, 256, 0, stream>>>(pts, vlc0, vlc1, vlc2,
                                                    pv0, pv1, pv2, sums,
                                                    vf012, feat, sWpA);
  k_vfgather<<<(NPTS + 255) / 256, 256, 0, stream>>>(pv0, pv1, pv2, vf012, feat);
  k_main<<<(NTILE + 31) / 32, 256, 0, stream>>>(pts, vlc3, vlc4, pv3, pv4,
                                                sums, feat, sWpO, vf34);
  k_scatter<<<15000, 256, 0, stream>>>(vf34, vfs3, vfs4, out);
}
